// GCNBranch_42700564857426
// MI455X (gfx1250) — compile-verified, matrix operand order checked by reading, not run on hardware
//
#include <hip/hip_runtime.h>
#include <math.h>

#define BB 128
#define NNODE 300
#define EE 300
#define DIN 768
#define DH 256
#define RR 26
#define NH 8
#define HD 32
#define QKV3 768          // 3*DH
#define EPAD 768          // padded sorted-edge slots per sample (16-aligned segments)
#define TILES 48          // EPAD/16
#define NPAD 320          // padded key length for ctx GEMM (300 -> 320)
#define MT_N 19           // ceil(300/16)
#define SROW 328          // LDS f32 score row stride (bank-conflict pad)
#define PROW 336          // LDS bf16 prob row stride (16B aligned, pad)

typedef __attribute__((ext_vector_type(16))) __bf16 v16bf;
typedef __attribute__((ext_vector_type(8)))  float  v8f;

struct FragB16 { union { v16bf v; uint4 q[2]; }; };

__device__ __forceinline__ unsigned short f2bf(float f) {
  unsigned int u = __float_as_uint(f);
  u += 0x7FFFu + ((u >> 16) & 1u);           // round-to-nearest-even
  return (unsigned short)(u >> 16);
}
__device__ __forceinline__ float bf2f(unsigned short h) {
  return __uint_as_float(((unsigned int)h) << 16);
}

// A/B fragment loader for v_wmma_f32_16x16x32_bf16.
// Source is a 16-row tile of 16-bit data, row-major with leading dim `ld`.
// Per ISA layout: lanes 0-15 hold row (lane&15), K = {k0..k0+7, k0+16..k0+23};
// lanes 16-31 hold same row, K = {k0+8..k0+15, k0+24..k0+31}.
__device__ __forceinline__ void load_frag(FragB16& f, const unsigned short* tile,
                                          int ld, int k0, int lane) {
  int r = lane & 15, hi = lane >> 4;
  const unsigned short* p = tile + (size_t)r * ld + k0 + hi * 8;
  f.q[0] = *(const uint4*)(p);
  f.q[1] = *(const uint4*)(p + 16);
}
__device__ __forceinline__ v8f wmma_bf16(const FragB16& a, const FragB16& b, v8f c) {
  return __builtin_amdgcn_wmma_f32_16x16x32_bf16(false, a.v, false, b.v,
                                                 (short)0, c, false, false);
}

// ---------------- conversion kernels ----------------
__global__ void k_cvt(const float* __restrict__ in, unsigned short* __restrict__ out, int n) {
  int i = blockIdx.x * blockDim.x + threadIdx.x;
  if (i < n) out[i] = f2bf(in[i]);
}
// W_rel [r][k][n] f32 -> [r][n][k] bf16
__global__ void k_cvt_wrel_t(const float* __restrict__ in, unsigned short* __restrict__ out, int n) {
  int i = blockIdx.x * blockDim.x + threadIdx.x;
  if (i >= n) return;
  int r = i / (DIN * DH);
  int rem = i - r * (DIN * DH);
  int k = rem / DH, c = rem - k * DH;
  out[((size_t)r * DH + c) * DIN + k] = f2bf(in[i]);
}
// W_loop [k][n] f32 -> [n][k] bf16
__global__ void k_cvt_wloop_t(const float* __restrict__ in, unsigned short* __restrict__ out, int n) {
  int i = blockIdx.x * blockDim.x + threadIdx.x;
  if (i >= n) return;
  int k = i / DH, c = i - k * DH;
  out[(size_t)c * DIN + k] = f2bf(in[i]);
}

// ---------------- edge bucketing (counting sort by relation, 16-aligned segments) ----------------
__global__ __launch_bounds__(256) void k_bucket(
    const int* __restrict__ etype, const int* __restrict__ src, const int* __restrict__ dst,
    int* __restrict__ srcS, int* __restrict__ dstS, int* __restrict__ tile_rel) {
  int b = blockIdx.x;
  __shared__ int cnt[RR], start[RR], cur[RR], rend[RR];
  int t = threadIdx.x;
  if (t < RR) cnt[t] = 0;
  __syncthreads();
  for (int e = t; e < EE; e += blockDim.x) atomicAdd(&cnt[etype[b * EE + e]], 1);
  __syncthreads();
  if (t == 0) {
    int c = 0;
    for (int r = 0; r < RR; ++r) {
      start[r] = c; cur[r] = c;
      int e2 = c + cnt[r];
      rend[r] = (e2 + 15) & ~15;
      c = rend[r];
    }
  }
  __syncthreads();
  for (int lt = t; lt < TILES; lt += blockDim.x) {
    int rowa = lt * 16, rr = -1;
    for (int r = 0; r < RR; ++r)
      if (cnt[r] > 0 && rowa >= start[r] && rowa < rend[r]) { rr = r; break; }
    tile_rel[b * TILES + lt] = rr;
  }
  for (int i = t; i < EPAD; i += blockDim.x) {
    srcS[b * EPAD + i] = -1;
    dstS[b * EPAD + i] = -1;
  }
  __syncthreads();
  for (int e = t; e < EE; e += blockDim.x) {
    int r = etype[b * EE + e];
    int pos = atomicAdd(&cur[r], 1);
    srcS[b * EPAD + pos] = src[b * EE + e];
    dstS[b * EPAD + pos] = dst[b * EE + e];
  }
}

// ---------------- relation-segmented message GEMM + scatter-add ----------------
// agg[b,dst] += gather(nf, src) @ W_rel[rel]   (bf16 WMMA, f32 accum, f32 atomics)
__global__ __launch_bounds__(256) void k_msg(
    const unsigned short* __restrict__ nf_bf,   // [B*NNODE][DIN]
    const unsigned short* __restrict__ wrel_t,  // [R][DH][DIN]
    const int* __restrict__ srcS, const int* __restrict__ dstS,
    const int* __restrict__ tile_rel,
    float* __restrict__ agg)                    // [B*NNODE][DH]
{
  int blk = blockIdx.x;
  int b = blk / TILES, lt = blk - b * TILES;
  int rel = tile_rel[b * TILES + lt];
  if (rel < 0) return;
  int tid = threadIdx.x, wave = tid >> 5, lane = tid & 31;
  int rfrag = lane & 15, hi = lane >> 4;
  int row0 = lt * 16;
  // Pad rows may load any valid source row: their WMMA output rows are
  // discarded via dst==-1 in the scatter epilogue, and row m of D depends
  // only on row m of A. Clamping avoids divergent zero-fill in the K-loop.
  int snode = srcS[b * EPAD + row0 + rfrag];
  if (snode < 0) snode = 0;
  const unsigned short* arow = nf_bf + ((size_t)b * NNODE + snode) * DIN;
  const unsigned short* wb = wrel_t + (size_t)rel * DH * DIN;
  int nc0 = wave * 32, nc1 = nc0 + 16;
  v8f c0 = {}, c1 = {};
  for (int k0 = 0; k0 < DIN; k0 += 32) {
    FragB16 a, b0, b1;
    const unsigned short* p = arow + k0 + hi * 8;
    a.q[0] = *(const uint4*)p;
    a.q[1] = *(const uint4*)(p + 16);
    load_frag(b0, wb + (size_t)nc0 * DIN, DIN, k0, lane);
    load_frag(b1, wb + (size_t)nc1 * DIN, DIN, k0, lane);
    c0 = wmma_bf16(a, b0, c0);
    c1 = wmma_bf16(a, b1, c1);
  }
  int dn[8];
  for (int j = 0; j < 8; ++j) dn[j] = dstS[b * EPAD + row0 + j + hi * 8];
  for (int j = 0; j < 8; ++j) {
    if (dn[j] >= 0) {
      float* d0 = &agg[((size_t)b * NNODE + dn[j]) * DH + (lane & 15)];
      atomicAdd(d0 + nc0, c0[j]);
      atomicAdd(d0 + nc1, c1[j]);
    }
  }
}

// ---------------- self-loop GEMM: h = agg + nf @ W_loop + b_rel ----------------
__global__ __launch_bounds__(256) void k_selfloop(
    const unsigned short* __restrict__ nf_bf,
    const unsigned short* __restrict__ wloop_t,   // [DH][DIN]
    const float* __restrict__ b_rel,
    float* __restrict__ h,                        // in: agg, out: h
    unsigned short* __restrict__ h_bf)
{
  int row0 = blockIdx.x * 16;
  int tid = threadIdx.x, wave = tid >> 5, lane = tid & 31;
  int hi = lane >> 4;
  int nc0 = wave * 32, nc1 = nc0 + 16;
  const unsigned short* atile = nf_bf + (size_t)row0 * DIN;
  v8f c0 = {}, c1 = {};
  for (int k0 = 0; k0 < DIN; k0 += 32) {
    FragB16 a, b0, b1;
    load_frag(a, atile, DIN, k0, lane);
    load_frag(b0, wloop_t + (size_t)nc0 * DIN, DIN, k0, lane);
    load_frag(b1, wloop_t + (size_t)nc1 * DIN, DIN, k0, lane);
    c0 = wmma_bf16(a, b0, c0);
    c1 = wmma_bf16(a, b1, c1);
  }
  for (int j = 0; j < 8; ++j) {
    int m = j + hi * 8;
    size_t base = (size_t)(row0 + m) * DH + (lane & 15);
    float v0 = h[base + nc0] + c0[j] + b_rel[nc0 + (lane & 15)];
    float v1 = h[base + nc1] + c1[j] + b_rel[nc1 + (lane & 15)];
    h[base + nc0] = v0; h_bf[base + nc0] = f2bf(v0);
    h[base + nc1] = v1; h_bf[base + nc1] = f2bf(v1);
  }
}

// ---------------- QKV GEMM: qkv = h @ in_proj_w^T + b; q pre-scaled; v transposed ----------------
__global__ __launch_bounds__(256) void k_qkv(
    const unsigned short* __restrict__ h_bf,     // [B*NNODE][DH]
    const unsigned short* __restrict__ wq_bf,    // in_proj_w bf16 [768][256] (already [e][d])
    const float* __restrict__ in_proj_b,
    unsigned short* __restrict__ qkv_bf,         // [B*NNODE][768] (q,k regions used)
    unsigned short* __restrict__ vT)             // [B][NH][HD][NPAD], zero-padded
{
  int row0 = blockIdx.x * 16;
  int tid = threadIdx.x, wave = tid >> 5, lane = tid & 31;
  int hi = lane >> 4;
  const unsigned short* atile = h_bf + (size_t)row0 * DH;
  v8f c[6] = {};
  for (int k0 = 0; k0 < DH; k0 += 32) {
    FragB16 a;
    load_frag(a, atile, DH, k0, lane);
    for (int nt = 0; nt < 6; ++nt) {
      int nc = (wave * 6 + nt) * 16;
      FragB16 bfr;
      load_frag(bfr, wq_bf + (size_t)nc * DH, DH, k0, lane);
      c[nt] = wmma_bf16(a, bfr, c[nt]);
    }
  }
  const float qscale = 0.17677669529663687f;  // 1/sqrt(HD)
  for (int nt = 0; nt < 6; ++nt) {
    int nc = (wave * 6 + nt) * 16;
    int e = nc + (lane & 15);
    float bias = in_proj_b[e];
    for (int j = 0; j < 8; ++j) {
      int row = row0 + j + hi * 8;
      float val = c[nt][j] + bias;
      if (e < 2 * DH) {
        if (e < DH) val *= qscale;
        qkv_bf[(size_t)row * QKV3 + e] = f2bf(val);
      } else {
        int ev = e - 2 * DH;
        int hh = ev >> 5, d = ev & 31;
        int bb = row / NNODE, nn = row - bb * NNODE;
        vT[(((size_t)bb * NH + hh) * HD + d) * NPAD + nn] = f2bf(val);
      }
    }
  }
}

// ---------------- fused attention per (b, head, 16-row q-tile); one wave ----------------
__global__ __launch_bounds__(32) void k_attn(
    const unsigned short* __restrict__ qkv_bf,
    const unsigned short* __restrict__ vT,
    unsigned short* __restrict__ ctx_bf)         // [B*NNODE][DH]
{
  int mt = blockIdx.x, hh = blockIdx.y, b = blockIdx.z;
  __shared__ float sc[16 * SROW];
  __shared__ unsigned short pr[16 * PROW];
  int lane = threadIdx.x;
  int rfrag = lane & 15, hi = lane >> 4;

  // Q fragment (K = HD = 32, single WMMA per key tile)
  int qrow = mt * 16 + rfrag; if (qrow > NNODE - 1) qrow = NNODE - 1;
  FragB16 a;
  {
    const unsigned short* p = qkv_bf + ((size_t)b * NNODE + qrow) * QKV3 + hh * HD + hi * 8;
    a.q[0] = *(const uint4*)p;
    a.q[1] = *(const uint4*)(p + 16);
  }
  // scores = q @ k^T (q pre-scaled)
  for (int nt = 0; nt < MT_N; ++nt) {
    int krow = nt * 16 + rfrag; if (krow > NNODE - 1) krow = NNODE - 1;
    FragB16 bfr;
    const unsigned short* kp =
        qkv_bf + ((size_t)b * NNODE + krow) * QKV3 + DH + hh * HD + hi * 8;
    bfr.q[0] = *(const uint4*)kp;
    bfr.q[1] = *(const uint4*)(kp + 16);
    v8f c = {};
    c = wmma_bf16(a, bfr, c);
    for (int j = 0; j < 8; ++j)
      sc[(j + hi * 8) * SROW + nt * 16 + (lane & 15)] = c[j];
  }
  __syncthreads();
  // row-wise softmax (lanes 0-15 each own a row), probs -> bf16, zero padding
  if (lane < 16) {
    int row = mt * 16 + lane;
    float* srow = &sc[lane * SROW];
    unsigned short* prow = &pr[lane * PROW];
    if (row < NNODE) {
      float mx = -1e30f;
      for (int i = 0; i < NNODE; ++i) mx = fmaxf(mx, srow[i]);
      float s = 0.f;
      for (int i = 0; i < NNODE; ++i) { float e = __expf(srow[i] - mx); srow[i] = e; s += e; }
      float inv = 1.f / s;
      for (int i = 0; i < NNODE; ++i) prow[i] = f2bf(srow[i] * inv);
      for (int i = NNODE; i < NPAD; ++i) prow[i] = 0;
    } else {
      for (int i = 0; i < NPAD; ++i) prow[i] = 0;
    }
  }
  __syncthreads();
  // ctx = attn @ v  (K = NPAD, vT pre-transposed [d][n])
  const unsigned short* vbase = vT + ((size_t)(b * NH + hh) * HD) * NPAD;
  for (int dt = 0; dt < 2; ++dt) {
    v8f c = {};
    for (int k0 = 0; k0 < NPAD; k0 += 32) {
      FragB16 a2, b2;
      const unsigned short* ap = &pr[(lane & 15) * PROW + k0 + hi * 8];
      a2.q[0] = *(const uint4*)ap;
      a2.q[1] = *(const uint4*)(ap + 16);
      const unsigned short* bp = vbase + (size_t)(dt * 16 + (lane & 15)) * NPAD + k0 + hi * 8;
      b2.q[0] = *(const uint4*)bp;
      b2.q[1] = *(const uint4*)(bp + 16);
      c = wmma_bf16(a2, b2, c);
    }
    for (int j = 0; j < 8; ++j) {
      int row = mt * 16 + j + hi * 8;
      if (row < NNODE)
        ctx_bf[((size_t)b * NNODE + row) * DH + hh * HD + dt * 16 + (lane & 15)] = f2bf(c[j]);
    }
  }
}

// ---------------- mean-pool ctx over nodes ----------------
__global__ void k_ctx_mean(const unsigned short* __restrict__ ctx_bf,
                           float* __restrict__ ctx_mean) {
  int i = blockIdx.x * blockDim.x + threadIdx.x;  // B*DH
  if (i >= BB * DH) return;
  int b = i / DH, c = i - b * DH;
  float s = 0.f;
  for (int n = 0; n < NNODE; ++n) s += bf2f(ctx_bf[((size_t)b * NNODE + n) * DH + c]);
  ctx_mean[i] = s * (1.0f / NNODE);
}

// ---------------- small fp32 linear: y[b][j] = x[b] . w[j] + bias[j] ----------------
__global__ void k_linear(const float* __restrict__ x, const float* __restrict__ w,
                         const float* __restrict__ bias, float* __restrict__ y, int K) {
  int b = blockIdx.x, j = threadIdx.x;
  const float* xr = x + (size_t)b * K;
  const float* wr = w + (size_t)j * K;
  float s = bias[j];
  for (int k = 0; k < K; ++k) s += xr[k] * wr[k];
  y[(size_t)b * blockDim.x + j] = s;
}

extern "C" void kernel_launch(void* const* d_in, const int* in_sizes, int n_in,
                              void* d_out, int out_size, void* d_ws, size_t ws_size,
                              hipStream_t stream) {
  const float* nf         = (const float*)d_in[0];
  const int*   src        = (const int*)d_in[1];
  const int*   dst        = (const int*)d_in[2];
  const int*   etype      = (const int*)d_in[3];
  const float* W_rel      = (const float*)d_in[4];
  const float* W_loop     = (const float*)d_in[5];
  const float* b_rel      = (const float*)d_in[6];
  const float* in_proj_w  = (const float*)d_in[7];
  const float* in_proj_b  = (const float*)d_in[8];
  const float* out_proj_w = (const float*)d_in[9];
  const float* out_proj_b = (const float*)d_in[10];
  const float* mlp_w      = (const float*)d_in[11];
  const float* mlp_b      = (const float*)d_in[12];
  float* out = (float*)d_out;

  char* w = (char*)d_ws;
  auto alloc = [&](size_t bytes) -> char* {
    char* p = w;
    w += (bytes + 255) & ~(size_t)255;
    return p;
  };
  unsigned short* nf_bf   = (unsigned short*)alloc((size_t)BB * NNODE * DIN * 2);
  unsigned short* wrel_t  = (unsigned short*)alloc((size_t)RR * DH * DIN * 2);
  unsigned short* wloop_t = (unsigned short*)alloc((size_t)DH * DIN * 2);
  unsigned short* wqkv_bf = (unsigned short*)alloc((size_t)QKV3 * DH * 2);
  int* srcS     = (int*)alloc((size_t)BB * EPAD * 4);
  int* dstS     = (int*)alloc((size_t)BB * EPAD * 4);
  int* tile_rel = (int*)alloc((size_t)BB * TILES * 4);
  float* h              = (float*)alloc((size_t)BB * NNODE * DH * 4);
  unsigned short* h_bf  = (unsigned short*)alloc((size_t)BB * NNODE * DH * 2);
  unsigned short* qkvb  = (unsigned short*)alloc((size_t)BB * NNODE * QKV3 * 2);
  unsigned short* vT    = (unsigned short*)alloc((size_t)BB * NH * HD * NPAD * 2);
  unsigned short* ctxb  = (unsigned short*)alloc((size_t)BB * NNODE * DH * 2);
  float* ctx_mean = (float*)alloc((size_t)BB * DH * 4);
  float* pooled   = (float*)alloc((size_t)BB * DH * 4);

  // bf16 conversions (+ transposes for B operands stored output-major)
  { int n = BB * NNODE * DIN; k_cvt<<<(n + 255) / 256, 256, 0, stream>>>(nf, nf_bf, n); }
  { int n = RR * DIN * DH;    k_cvt_wrel_t<<<(n + 255) / 256, 256, 0, stream>>>(W_rel, wrel_t, n); }
  { int n = DIN * DH;         k_cvt_wloop_t<<<(n + 255) / 256, 256, 0, stream>>>(W_loop, wloop_t, n); }
  { int n = QKV3 * DH;        k_cvt<<<(n + 255) / 256, 256, 0, stream>>>(in_proj_w, wqkv_bf, n); }

  k_bucket<<<BB, 256, 0, stream>>>(etype, src, dst, srcS, dstS, tile_rel);
  hipMemsetAsync(h, 0, (size_t)BB * NNODE * DH * 4, stream);
  hipMemsetAsync(vT, 0, (size_t)BB * NH * HD * NPAD * 2, stream);

  k_msg<<<BB * TILES, 256, 0, stream>>>(nf_bf, wrel_t, srcS, dstS, tile_rel, h);
  k_selfloop<<<(BB * NNODE) / 16, 256, 0, stream>>>(nf_bf, wloop_t, b_rel, h, h_bf);
  k_qkv<<<(BB * NNODE) / 16, 256, 0, stream>>>(h_bf, wqkv_bf, in_proj_b, qkvb, vT);
  k_attn<<<dim3(MT_N, NH, BB), 32, 0, stream>>>(qkvb, vT, ctxb);
  { int n = BB * DH; k_ctx_mean<<<(n + 255) / 256, 256, 0, stream>>>(ctxb, ctx_mean); }
  k_linear<<<BB, DH, 0, stream>>>(ctx_mean, out_proj_w, out_proj_b, pooled, DH);
  k_linear<<<BB, DH, 0, stream>>>(pooled, mlp_w, mlp_b, out, DH);
}
